// VisionModelMoe_42554535968926
// MI455X (gfx1250) — matrix operand order repaired and checked by reading, not compiled, not run on hardware
//
#include <hip/hip_runtime.h>
#include <hip/hip_bf16.h>

// ---------------------------------------------------------------------------
// MoE vision block, top-2 routed, bf16 WMMA expert FFN for gfx1250 (MI455X).
// B=32, C=D=128, H=W=32 -> T=32768 tokens, E=16 experts, K=2, HID=512.
// FFN: 4 token-tiles per block (1/wave), B operands staged in LDS with
// double-buffered async global->LDS copies (4x weight-traffic reduction).
// ---------------------------------------------------------------------------

#define NTOK   32768
#define NEXP   16
#define DIMD   128
#define HIDN   512
#define NASSGN (NTOK * 2)

typedef __bf16 bf16;
typedef __attribute__((ext_vector_type(8)))  bf16  v8bf;
typedef __attribute__((ext_vector_type(16))) bf16  v16bf;
typedef __attribute__((ext_vector_type(8)))  float v8f;
typedef int v4i __attribute__((vector_size(16)));

static __device__ __forceinline__ v16bf mk16(v8bf lo, v8bf hi) {
  return __builtin_shufflevector(lo, hi, 0,1,2,3,4,5,6,7,8,9,10,11,12,13,14,15);
}
// A-operand (16x32 bf16): lane half h: elems0-15=K[kb..kb+15] (16 consecutive K)
static __device__ __forceinline__ v16bf loadA(const bf16* p) {
  v8bf lo = *(const v8bf*)p;
  v8bf hi = *(const v8bf*)(p + 8);
  return mk16(lo, hi);
}
// B-operand (32x16 bf16 from transposed weights): 16 consecutive K per lane
static __device__ __forceinline__ v16bf loadB(const bf16* p) {
  v8bf lo = *(const v8bf*)p;
  v8bf hi = *(const v8bf*)(p + 8);
  return mk16(lo, hi);
}

// ---- async global->LDS staging (ASYNCcnt path), with safe fallback --------
#if __has_builtin(__builtin_amdgcn_global_load_async_to_lds_b128)
#define CP_ASYNC 1
#endif

static __device__ __forceinline__ void cp16B(bf16* l, const bf16* g) {
#ifdef CP_ASYNC
  __builtin_amdgcn_global_load_async_to_lds_b128(
      (__attribute__((address_space(1))) v4i*)g,
      (__attribute__((address_space(3))) v4i*)l, 0, 0);
#else
  *(v8bf*)l = *(const v8bf*)g;
#endif
}
static __device__ __forceinline__ void cp_wait() {
#ifdef CP_ASYNC
#if __has_builtin(__builtin_amdgcn_s_wait_asynccnt)
  __builtin_amdgcn_s_wait_asynccnt(0);
#else
  asm volatile("s_wait_asynccnt 0x0" ::: "memory");
#endif
#endif
}

// ------------------------------- utility kernels ---------------------------

__global__ void k_zero_counts(int* counts) {
  if (threadIdx.x < NEXP) counts[threadIdx.x] = 0;
}

// src: Z matrices [M][N] f32 -> dst: Z matrices [N][M] bf16. M,N multiples of 32.
__global__ __launch_bounds__(256) void k_transpose_bf16(
    const float* __restrict__ src, bf16* __restrict__ dst, int M, int N) {
  __shared__ float tile[32][33];
  const int z  = blockIdx.z;
  const float* s = src + (size_t)z * M * N;
  bf16*        d = dst + (size_t)z * M * N;
  const int n0 = blockIdx.x * 32, m0 = blockIdx.y * 32;
  const int tx = threadIdx.x & 31, ty = threadIdx.x >> 5;
#pragma unroll
  for (int i = 0; i < 4; ++i)
    tile[ty + i * 8][tx] = s[(size_t)(m0 + ty + i * 8) * N + n0 + tx];
  __syncthreads();
#pragma unroll
  for (int i = 0; i < 4; ++i)
    d[(size_t)(n0 + ty + i * 8) * M + m0 + tx] = (bf16)tile[tx][ty + i * 8];
}

// ------------------------------- gating ------------------------------------

static __device__ __forceinline__ unsigned pcg(unsigned v) {
  v = v * 747796405u + 2891336453u;
  unsigned w = ((v >> ((v >> 28) + 4u)) ^ v) * 277803737u;
  return (w >> 22) ^ w;
}
// Deterministic stand-in for jax.random.normal(key 42) (Box-Muller on a hash).
static __device__ __forceinline__ float gauss(unsigned id) {
  unsigned a = pcg(id * 2u + 1u), b = pcg(id * 2u + 2u);
  float u1 = ((a >> 8) + 0.5f) * (1.0f / 16777216.0f);
  float u2 = ((b >> 8) + 0.5f) * (1.0f / 16777216.0f);
  return sqrtf(-2.0f * logf(u1)) * cosf(6.28318530718f * u2);
}

__global__ __launch_bounds__(256) void k_gating(
    const float* __restrict__ x,
    const float* __restrict__ gw, const float* __restrict__ gb,
    const float* __restrict__ nw, const float* __restrict__ nb,
    float* __restrict__ gates, int* __restrict__ tok_e,
    float* __restrict__ tok_p, int* __restrict__ counts) {
  const int t  = blockIdx.x * 256 + threadIdx.x;
  const int b  = t >> 10, hw = t & 1023;
  const float* xp = x + (size_t)b * (DIMD * 1024) + hw;

  float g[NEXP], nz[NEXP];
#pragma unroll
  for (int e = 0; e < NEXP; ++e) { g[e] = gb[e]; nz[e] = nb[e]; }
  for (int c = 0; c < DIMD; ++c) {
    const float xv = xp[(size_t)c * 1024];
#pragma unroll
    for (int e = 0; e < NEXP; ++e) {
      g[e]  = fmaf(xv, gw[c * NEXP + e], g[e]);
      nz[e] = fmaf(xv, nw[c * NEXP + e], nz[e]);
    }
  }
#pragma unroll
  for (int e = 0; e < NEXP; ++e) {
    const float sp = nz[e] > 20.f ? nz[e] : log1pf(expf(nz[e]));
    g[e] += gauss((unsigned)(t * NEXP + e)) * sp;
  }
  float v0 = -3.4e38f, v1 = -3.4e38f; int i0 = 0, i1 = 0;
#pragma unroll
  for (int e = 0; e < NEXP; ++e) {
    if (g[e] > v0)      { v1 = v0; i1 = i0; v0 = g[e]; i0 = e; }
    else if (g[e] > v1) { v1 = g[e]; i1 = e; }
  }
  const float e1 = expf(v1 - v0);
  const float p0 = 1.0f / (1.0f + e1);
  const float p1 = e1 / (1.0f + e1);
#pragma unroll
  for (int e = 0; e < NEXP; ++e)
    gates[(size_t)t * NEXP + e] = (e == i0) ? p0 : ((e == i1) ? p1 : 0.0f);
  tok_e[2 * t] = i0; tok_e[2 * t + 1] = i1;
  tok_p[2 * t] = p0; tok_p[2 * t + 1] = p1;
  atomicAdd(&counts[i0], 1);
  atomicAdd(&counts[i1], 1);
}

__global__ void k_scan(const int* __restrict__ counts,
                       int* __restrict__ offsets, int* __restrict__ cursor) {
  if (threadIdx.x == 0) {
    int acc = 0;
    for (int e = 0; e < NEXP; ++e) { offsets[e] = acc; cursor[e] = acc; acc += counts[e]; }
    offsets[NEXP] = acc;
  }
}

__global__ __launch_bounds__(256) void k_scatter(
    const int* __restrict__ tok_e, int* __restrict__ cursor,
    int* __restrict__ bucket_tok, int* __restrict__ assign_pos) {
  const int i = blockIdx.x * 256 + threadIdx.x;
  if (i >= NASSGN) return;
  const int e   = tok_e[i];
  const int pos = atomicAdd(&cursor[e], 1);
  bucket_tok[pos] = i >> 1;
  assign_pos[i]   = pos;
}

// ------------------------- expert FFN (bf16 WMMA) --------------------------
// Block = 128 threads (4 waves) = 4 token-tiles of one expert (1 tile/wave).
// B operands staged in LDS (double-buffered, async): every weight byte is
// loaded once per block and feeds 4 tiles' WMMAs.
// LDS: hdn 4*16*512 bf16 (64KB, per-wave private) + 2*8KB staging = 80KB.

#define SMEM_BYTES (4 * 16 * HIDN * 2 + 2 * 4096 * 2)

__global__ __launch_bounds__(128) void k_ffn(
    const bf16*  __restrict__ xb,       // [T, D]
    const bf16*  __restrict__ w1t,      // [E, HID, D]
    const bf16*  __restrict__ w2t,      // [E, D, HID]
    const float* __restrict__ b1, const float* __restrict__ b2,
    const int*   __restrict__ counts, const int* __restrict__ offsets,
    const int*   __restrict__ bucket_tok,
    float*       __restrict__ eo_buf) { // [2T, D]
  const int e   = blockIdx.y;
  const int cnt = counts[e];
  if ((int)blockIdx.x * 64 >= cnt) return;            // uniform early-exit
  const int base0 = offsets[e];

  extern __shared__ bf16 smem[];
  bf16* hdn = smem;                                   // [4][16][512]
  bf16* bst = smem + 4 * 16 * HIDN;                   // 2 x 4096 elems

  const int tid  = threadIdx.x;
  const int lane = tid & 31, wave = tid >> 5;
  const int r    = lane & 15, hi = lane >> 4;

  const int tstart = blockIdx.x * 64 + wave * 16;     // this wave's tile
  const int gidx   = min(tstart + r, cnt - 1);
  const int tok    = bucket_tok[base0 + gidx];

  // ---- A operand for this wave's tile -----------------------------------
  const bf16* ap = xb + (size_t)tok * DIMD;
  v16bf A[4];
#pragma unroll
  for (int kc = 0; kc < 4; ++kc) A[kc] = loadA(ap + kc * 32 + hi * 16);

  const bf16* w1e = w1t + (size_t)e * HIDN * DIMD;    // [HID][D]
  const bf16* w2e = w2t + (size_t)e * DIMD * HIDN;    // [D][HID]
  bf16* hw_ = hdn + wave * 16 * HIDN;                 // per-wave hdn region

  // -------------------------------- GEMM1 --------------------------------
  const int s1row = tid >> 3, s1seg = (tid & 7) * 16; // 128 thr copy 16x128
  {
    const bf16* g0 = w1e + (size_t)s1row * DIMD + s1seg;
    cp16B(bst + s1row * DIMD + s1seg, g0);
    cp16B(bst + s1row * DIMD + s1seg + 8, g0 + 8);
  }
  cp_wait();
  __syncthreads();
  for (int nt = 0; nt < HIDN / 16; ++nt) {
    const int cur = nt & 1;
    if (nt + 1 < HIDN / 16) {                         // prefetch next B tile
      const bf16* g = w1e + (size_t)((nt + 1) * 16 + s1row) * DIMD + s1seg;
      bf16* l = bst + (cur ^ 1) * 4096 + s1row * DIMD + s1seg;
      cp16B(l, g);
      cp16B(l + 8, g + 8);
    }
    const bf16* bb = bst + cur * 4096;
    v8f acc = {};
#pragma unroll
    for (int kc = 0; kc < 4; ++kc)
      acc = __builtin_amdgcn_wmma_f32_16x16x32_bf16(
          false, A[kc], false, loadB(bb + r * DIMD + kc * 32 + hi * 16),
          (short)0, acc, false, false);
    const int   col  = nt * 16 + r;
    const float bias = b1[e * HIDN + col];
#pragma unroll
    for (int i = 0; i < 8; ++i) {
      const int row = i + hi * 8;
      float v = acc[i] + bias;
      hw_[row * HIDN + col] = (bf16)(v > 0.0f ? v : 0.0f);  // relu -> bf16
    }
    cp_wait();
    __syncthreads();
  }

  // -------------------------------- GEMM2 --------------------------------
  v8f acc2[8] = {};
  const bf16* g2base = w2e + (size_t)tid * HIDN;      // thread copies row d=tid
  {
#pragma unroll
    for (int j = 0; j < 4; ++j) cp16B(bst + tid * 32 + j * 8, g2base + j * 8);
  }
  cp_wait();
  __syncthreads();
  for (int kc = 0; kc < 16; ++kc) {
    const int cur = kc & 1;
    if (kc + 1 < 16) {                                // prefetch next k-slice
      const bf16* g = g2base + (kc + 1) * 32;
      bf16* l = bst + (cur ^ 1) * 4096 + tid * 32;
#pragma unroll
      for (int j = 0; j < 4; ++j) cp16B(l + j * 8, g + j * 8);
    }
    const bf16* bb = bst + cur * 4096;
    v16bf Av = loadA(hw_ + r * HIDN + kc * 32 + hi * 16);
#pragma unroll
    for (int dnt = 0; dnt < 8; ++dnt)
      acc2[dnt] = __builtin_amdgcn_wmma_f32_16x16x32_bf16(
          false, Av, false, loadB(bb + (dnt * 16 + r) * 32 + hi * 16),
          (short)0, acc2[dnt], false, false);
    cp_wait();
    __syncthreads();
  }
  // epilogue: +b2, scatter rows of this wave's tile
#pragma unroll
  for (int dnt = 0; dnt < 8; ++dnt) {
    const int   col  = dnt * 16 + r;
    const float bias = b2[e * DIMD + col];
#pragma unroll
    for (int i = 0; i < 8; ++i) {
      const int row = i + hi * 8;
      if (tstart + row < cnt)
        eo_buf[(size_t)(base0 + tstart + row) * DIMD + col] = acc2[dnt][i] + bias;
    }
  }
}

// ------------------------------- combine -----------------------------------
__global__ __launch_bounds__(256) void k_combine(
    const float* __restrict__ eo_buf, const int* __restrict__ assign_pos,
    const float* __restrict__ tok_p, float* __restrict__ out) {
  const int idx = blockIdx.x * 256 + threadIdx.x;     // 2^22 elements
  const int hw = idx & 1023;
  const int c  = (idx >> 10) & 127;
  const int b  = idx >> 17;
  const int t  = (b << 10) | hw;
  const int p0 = assign_pos[2 * t], p1 = assign_pos[2 * t + 1];
  out[idx] = tok_p[2 * t]     * eo_buf[(size_t)p0 * DIMD + c]
           + tok_p[2 * t + 1] * eo_buf[(size_t)p1 * DIMD + c];
}

// ------------------------------- launcher ----------------------------------

extern "C" void kernel_launch(void* const* d_in, const int* in_sizes, int n_in,
                              void* d_out, int out_size, void* d_ws, size_t ws_size,
                              hipStream_t stream) {
  (void)in_sizes; (void)n_in; (void)out_size; (void)ws_size;
  const float* x  = (const float*)d_in[0];
  const float* gw = (const float*)d_in[1];
  const float* gb = (const float*)d_in[2];
  const float* nw = (const float*)d_in[3];
  const float* nb = (const float*)d_in[4];
  const float* w1 = (const float*)d_in[5];
  const float* b1 = (const float*)d_in[6];
  const float* w2 = (const float*)d_in[7];
  const float* b2 = (const float*)d_in[8];
  // d_in[9] = top_k (always 2; hardcoded)

  float* out   = (float*)d_out;
  float* gates = (float*)d_out + (size_t)32 * 128 * 32 * 32;

  char* ws = (char*)d_ws;
  bf16*  xb         = (bf16*)(ws + 0);                // 8 MB  [T,D]
  bf16*  w1t        = (bf16*)(ws + 8388608);          // 2 MB  [E,HID,D]
  bf16*  w2t        = (bf16*)(ws + 10485760);         // 2 MB  [E,D,HID]
  float* eo_buf     = (float*)(ws + 12582912);        // 32 MB [2T,D]
  int*   tok_e      = (int*)  (ws + 46137344);
  float* tok_p      = (float*)(ws + 46399488);
  int*   bucket_tok = (int*)  (ws + 46661632);
  int*   assign_pos = (int*)  (ws + 46923776);
  int*   counts     = (int*)  (ws + 47185920);
  int*   offsets    = (int*)  (ws + 47186048);
  int*   cursor     = (int*)  (ws + 47186176);

  k_zero_counts<<<1, 32, 0, stream>>>(counts);

  k_transpose_bf16<<<dim3(32, 4, 32), 256, 0, stream>>>(x, xb, 128, 1024);
  k_transpose_bf16<<<dim3(16, 4, 16), 256, 0, stream>>>(w1, w1t, 128, 512);
  k_transpose_bf16<<<dim3(4, 16, 16), 256, 0, stream>>>(w2, w2t, 512, 128);

  k_gating<<<NTOK / 256, 256, 0, stream>>>(x, gw, gb, nw, nb,
                                           gates, tok_e, tok_p, counts);
  k_scan<<<1, 32, 0, stream>>>(counts, offsets, cursor);
  k_scatter<<<NASSGN / 256, 256, 0, stream>>>(tok_e, cursor, bucket_tok, assign_pos);

  // 4 tiles/block; worst case 2048 tiles -> 512 blocks per expert
  k_ffn<<<dim3(512, NEXP), 128, SMEM_BYTES, stream>>>(
      xb, w1t, w2t, b1, b2, counts, offsets, bucket_tok, eo_buf);

  k_combine<<<(32 * 128 * 1024) / 256, 256, 0, stream>>>(eo_buf, assign_pos, tok_p, out);
}
